// ResnetBlock_8512625180760
// MI455X (gfx1250) — compile-verified, matrix operand order checked by reading, not run on hardware
//
#include <hip/hip_runtime.h>
#include <hip/hip_bf16.h>

typedef __bf16 bf16;
typedef __attribute__((ext_vector_type(16))) bf16  v16bf;
typedef __attribute__((ext_vector_type(8)))  bf16  v8bf;
typedef __attribute__((ext_vector_type(8)))  float v8f;

union Frag16 { v16bf v; v8bf h[2]; };

#define KNN_K 16

// ---------------------------------------------------------------------------
// KNN: one thread per center; LDS-tiled point positions (broadcast reads);
// register top-16 via unrolled insertion (ascending d2, stable ties -> matches
// jax.lax.top_k(-d2) ordering). Also emits gathered center positions.
// ---------------------------------------------------------------------------
__global__ __launch_bounds__(256)
void knn_kernel(const float* __restrict__ pos, int n_in, int n_out, int stride,
                int* __restrict__ knn, float* __restrict__ cpos)
{
    __shared__ float tile[256 * 3];
    const int b = blockIdx.y;
    const int j = blockIdx.x * 256 + threadIdx.x;
    const float* pb = pos + (long)b * n_in * 3;
    const int src = j * stride;
    const float cx = pb[src * 3 + 0];
    const float cy = pb[src * 3 + 1];
    const float cz = pb[src * 3 + 2];

    float dist[KNN_K];
    int   idxr[KNN_K];
#pragma unroll
    for (int t = 0; t < KNN_K; ++t) { dist[t] = 3.4e38f; idxr[t] = 0; }

    for (int base = 0; base < n_in; base += 256) {
        for (int e = threadIdx.x; e < 256 * 3; e += 256)
            tile[e] = pb[base * 3 + e];
        __syncthreads();
#pragma unroll 4
        for (int p = 0; p < 256; ++p) {
            float dx = tile[p * 3 + 0] - cx;
            float dy = tile[p * 3 + 1] - cy;
            float dz = tile[p * 3 + 2] - cz;
            float d2 = dx * dx + dy * dy + dz * dz;
            if (d2 < dist[KNN_K - 1]) {
                float dd = d2; int ii = base + p;
#pragma unroll
                for (int t = 0; t < KNN_K; ++t) {
                    if (dd < dist[t]) {
                        float td = dist[t]; int ti = idxr[t];
                        dist[t] = dd; idxr[t] = ii; dd = td; ii = ti;
                    }
                }
            }
        }
        __syncthreads();
    }

#pragma unroll
    for (int t = 0; t < KNN_K; ++t)
        knn[((long)b * n_out + j) * KNN_K + t] = idxr[t];
    cpos[((long)b * n_out + j) * 3 + 0] = cx;
    cpos[((long)b * n_out + j) * 3 + 1] = cy;
    cpos[((long)b * n_out + j) * 3 + 2] = cz;
}

// ---------------------------------------------------------------------------
// Pack weights into a WMMA-B-fragment-ready bf16 layout:
//   Bp[(t*kchunks + kc)*512 + lane*16 + i]
// lane: col = lane&15 (N), hi = lane>>4 selects K sub-block
// i<8  -> K = kc*32 + hi*8 + i ; i>=8 -> K = kc*32 + 16 + hi*8 + (i-8)
// transposed==0: W is [128][16][CIN], B[K][N] = W[N][K/CIN][K%CIN]
// transposed==1: W is [K][128],       B[K][N] = W[K][N]
// ---------------------------------------------------------------------------
__global__ __launch_bounds__(256)
void pack_w_kernel(const float* __restrict__ W, bf16* __restrict__ Bp,
                   int CIN, int KDIM, int transposed)
{
    const int e = blockIdx.x * 256 + threadIdx.x;
    const int total = 128 * KDIM;
    if (e >= total) return;
    const int i    = e & 15;
    const int lane = (e >> 4) & 31;
    const int rest = e >> 9;
    const int kchunks = KDIM >> 5;
    const int kc = rest % kchunks;
    const int t  = rest / kchunks;
    const int col = lane & 15;
    const int hi  = lane >> 4;
    const int N = t * 16 + col;
    const int K = kc * 32 + (i < 8 ? hi * 8 + i : 16 + hi * 8 + (i - 8));
    float v;
    if (transposed) {
        v = W[K * 128 + N];
    } else {
        int k = K / CIN, c = K % CIN;
        v = W[((long)N * 16 + k) * CIN + c];
    }
    Bp[e] = (bf16)v;
}

__global__ __launch_bounds__(256)
void fill_idx_kernel(int* __restrict__ nbr, int n_out, int stride, int total)
{
    int e = blockIdx.x * 256 + threadIdx.x;
    if (e < total) nbr[e] = (e % n_out) * stride;
}

__global__ __launch_bounds__(256)
void cvt_bf16_kernel(const float* __restrict__ x, bf16* __restrict__ y, int total)
{
    int e = blockIdx.x * 256 + threadIdx.x;
    if (e < total) y[e] = (bf16)x[e];
}

// ---------------------------------------------------------------------------
// Gather-GEMM via bf16 WMMA.
//   out[b,j,o] = bias[o] + sum_{k,c} ch[b, nbr[j,k], c] * W[o,k,c]
// Block = 256 thr (8 waves): 32 centers x 128 out channels.
// Each wave owns one 16-wide N tile and keeps TWO 16x16 accumulators
// (M tiles 0-15 and 16-31) so every B fragment feeds two WMMAs.
// Activations are pre-converted bf16; staging is pure 16-byte copies.
// ---------------------------------------------------------------------------
__global__ __launch_bounds__(256)
void gconv_wmma(const bf16* __restrict__ ch, const int* __restrict__ nbr,
                const bf16* __restrict__ Bp, const float* __restrict__ bias,
                float* __restrict__ out, int n_in, int n_out, int CIN, int KN)
{
    extern __shared__ bf16 lds[];
    const int KDIM  = KN * CIN;
    const int SLICE = KDIM < 512 ? KDIM : 512;   // 32 rows x 512 -> 33 KB LDS
    const int rstride = SLICE + 8;               // bank-conflict padding
    const int b    = blockIdx.y;
    const int row0 = blockIdx.x * 32;
    const int tid  = threadIdx.x;
    const int wave = tid >> 5;
    const int lane = tid & 31;
    const int col  = lane & 15;   // A row (M) / B-D column (N) within tile
    const int hi   = lane >> 4;   // K sub-block select

    v8f acc0, acc1;
    const float bv = bias[wave * 16 + col];
#pragma unroll
    for (int i = 0; i < 8; ++i) { acc0[i] = bv; acc1[i] = bv; }

    const int kchunksTotal = KDIM >> 5;
    const bf16* bp = Bp + ((long)wave * kchunksTotal) * 512 + (long)lane * 16;
    const bf16* arow0 = lds + (long)col * rstride + hi * 8;
    const bf16* arow1 = arow0 + 16 * rstride;

    const int CH8 = SLICE >> 3;                  // 8-element copy chunks per row
    for (int s = 0; s < KDIM; s += SLICE) {
        // stage gathered bf16 activations: 32 rows x SLICE, 16B granules
        for (int e = tid; e < 32 * CH8; e += 256) {
            int r  = e / CH8;
            int k8 = e - r * CH8;
            int kg = s + k8 * 8;
            int k  = kg / CIN;
            int c  = kg - k * CIN;
            int nb = nbr[((long)b * n_out + row0 + r) * KN + k];
            *(v8bf*)(lds + r * rstride + k8 * 8) =
                *(const v8bf*)(ch + ((long)b * n_in + nb) * CIN + c);
        }
        __syncthreads();

        const int kchunks = SLICE >> 5;
#pragma unroll 4
        for (int kc = 0; kc < kchunks; ++kc) {
            const bf16* bcur = bp + (long)kc * 512;
            __builtin_prefetch(bcur + 1024, 0, 1);   // 2 chunks ahead (B stream)
            Frag16 a0, a1, bb;
            bb.h[0] = *(const v8bf*)(bcur);
            bb.h[1] = *(const v8bf*)(bcur + 8);
            a0.h[0] = *(const v8bf*)(arow0 + kc * 32);
            a0.h[1] = *(const v8bf*)(arow0 + kc * 32 + 16);
            a1.h[0] = *(const v8bf*)(arow1 + kc * 32);
            a1.h[1] = *(const v8bf*)(arow1 + kc * 32 + 16);
            acc0 = __builtin_amdgcn_wmma_f32_16x16x32_bf16(
                false, a0.v, false, bb.v, (short)0, acc0, false, false);
            acc1 = __builtin_amdgcn_wmma_f32_16x16x32_bf16(
                false, a1.v, false, bb.v, (short)0, acc1, false, false);
        }
        __syncthreads();
        bp += (long)kchunks * 512;
    }

    // D layout: VGPR r -> M = r + 8*hi, N = col
#pragma unroll
    for (int r = 0; r < 8; ++r) {
        int m = r + hi * 8;
        out[((long)b * n_out + row0 + m) * 128 + wave * 16 + col]      = acc0[r];
        out[((long)b * n_out + row0 + 16 + m) * 128 + wave * 16 + col] = acc1[r];
    }
}

// ---------------------------------------------------------------------------
// LayerNorm(128) + SiLU, optional residual add (res row = b*res_n + j*resMul).
// One wave32 per row. Writes f32 (yf) and/or bf16 (yb) outputs.
// ---------------------------------------------------------------------------
__global__ __launch_bounds__(256)
void ln_silu_kernel(const float* __restrict__ x, const float* __restrict__ g,
                    const float* __restrict__ beta, const float* __restrict__ res,
                    int resMul, int res_n, int n_out,
                    float* __restrict__ yf, bf16* __restrict__ yb, int nrows)
{
    const int row  = blockIdx.x * 8 + (threadIdx.x >> 5);
    const int lane = threadIdx.x & 31;
    if (row >= nrows) return;
    const float* xr = x + (long)row * 128;
    float v[4], s = 0.f, s2 = 0.f;
#pragma unroll
    for (int i = 0; i < 4; ++i) {
        v[i] = xr[lane + i * 32];
        s  += v[i];
        s2 += v[i] * v[i];
    }
#pragma unroll
    for (int m = 16; m >= 1; m >>= 1) {
        s  += __shfl_xor(s,  m, 32);
        s2 += __shfl_xor(s2, m, 32);
    }
    const float mu  = s * (1.0f / 128.0f);
    const float var = s2 * (1.0f / 128.0f) - mu * mu;
    const float rs  = rsqrtf(var + 1e-5f);

    const float* rr = nullptr;
    if (res) {
        int jb = row % n_out, bb = row / n_out;
        rr = res + ((long)bb * res_n + (long)jb * resMul) * 128;
    }
#pragma unroll
    for (int i = 0; i < 4; ++i) {
        int c = lane + i * 32;
        float t  = (v[i] - mu) * rs * g[c] + beta[c];
        float sl = t / (1.0f + __expf(-t));
        if (rr) sl += rr[c];
        if (yf) yf[(long)row * 128 + c] = sl;
        if (yb) yb[(long)row * 128 + c] = (bf16)sl;
    }
}

// ---------------------------------------------------------------------------
extern "C" void kernel_launch(void* const* d_in, const int* in_sizes, int n_in_cnt,
                              void* d_out, int out_size, void* d_ws, size_t ws_size,
                              hipStream_t stream)
{
    const float* pos  = (const float*)d_in[0];
    const float* ch   = (const float*)d_in[1];
    const float* W1   = (const float*)d_in[2];
    const float* b1   = (const float*)d_in[3];
    const float* Wres = (const float*)d_in[4];
    const float* bres = (const float*)d_in[5];
    const float* W2   = (const float*)d_in[6];
    const float* b2   = (const float*)d_in[7];
    const float* g1   = (const float*)d_in[8];
    const float* be1  = (const float*)d_in[9];
    const float* g2   = (const float*)d_in[10];
    const float* be2  = (const float*)d_in[11];

    const int B = 2, N0 = 8192, N1 = 4096, N2 = 2048;

    char* w = (char*)d_ws;
    auto alloc = [&](size_t bytes) {
        char* p = w; w += (bytes + 255) & ~(size_t)255; return p;
    };
    int*   knn1   = (int*)  alloc((size_t)B * N1 * 16 * 4);
    float* cpos1  = (float*)alloc((size_t)B * N1 * 3  * 4);
    float* conv1  = (float*)alloc((size_t)B * N1 * 128 * 4);
    float* res1   = (float*)alloc((size_t)B * N1 * 128 * 4);
    bf16*  act1bf = (bf16*) alloc((size_t)B * N1 * 128 * 2);
    int*   knn2   = (int*)  alloc((size_t)B * N2 * 16 * 4);
    float* conv2  = (float*)alloc((size_t)B * N2 * 128 * 4);
    bf16*  Bp1    = (bf16*) alloc((size_t)128 * 1024 * 2);
    bf16*  BpR    = (bf16*) alloc((size_t)128 * 64   * 2);
    bf16*  Bp2    = (bf16*) alloc((size_t)128 * 2048 * 2);
    int*   cidx1  = (int*)  alloc((size_t)B * N1 * 4);
    bf16*  chbf   = (bf16*) alloc((size_t)B * N0 * 64 * 2);

    float* out_pos = (float*)d_out;                        // [B, N2, 3]
    float* out_ch  = (float*)d_out + (size_t)B * N2 * 3;   // [B, N2, 128]

    // prep: weight packing + activation bf16 conversion + center indices
    pack_w_kernel<<<(128 * 1024 + 255) / 256, 256, 0, stream>>>(W1,   Bp1, 64,  1024, 0);
    pack_w_kernel<<<(128 * 64   + 255) / 256, 256, 0, stream>>>(Wres, BpR, 64,  64,   1);
    pack_w_kernel<<<(128 * 2048 + 255) / 256, 256, 0, stream>>>(W2,   Bp2, 128, 2048, 0);
    fill_idx_kernel<<<(B * N1 + 255) / 256, 256, 0, stream>>>(cidx1, N1, 2, B * N1);
    cvt_bf16_kernel<<<(B * N0 * 64 + 255) / 256, 256, 0, stream>>>(ch, chbf, B * N0 * 64);

    // layer 1
    knn_kernel<<<dim3(N1 / 256, B), 256, 0, stream>>>(pos, N0, N1, 2, knn1, cpos1);
    gconv_wmma<<<dim3(N1 / 32, B), 256, 32 * (512 + 8) * 2, stream>>>(
        chbf, knn1, Bp1, b1, conv1, N0, N1, 64, 16);
    gconv_wmma<<<dim3(N1 / 32, B), 256, 32 * (64 + 8) * 2, stream>>>(
        chbf, cidx1, BpR, bres, res1, N0, N1, 64, 1);
    ln_silu_kernel<<<(B * N1) / 8, 256, 0, stream>>>(
        conv1, g1, be1, nullptr, 0, 0, N1, nullptr, act1bf, B * N1);

    // layer 2 (center positions written straight into d_out's pos region)
    knn_kernel<<<dim3(N2 / 256, B), 256, 0, stream>>>(cpos1, N1, N2, 2, knn2, out_pos);
    gconv_wmma<<<dim3(N2 / 32, B), 256, 32 * (512 + 8) * 2, stream>>>(
        act1bf, knn2, Bp2, b2, conv2, N1, N2, 128, 16);
    ln_silu_kernel<<<(B * N2) / 8, 256, 0, stream>>>(
        conv2, g2, be2, res1, 2, N1, N2, out_ch, nullptr, B * N2);
}
